// SparseLSTMCell_23940147708541
// MI455X (gfx1250) — matrix operand order repaired and checked by reading, not compiled, not run on hardware
//
#include <hip/hip_runtime.h>
#include <cstdint>

// ---------------------------------------------------------------------------
// SparseLSTMCell forward for MI455X (gfx1250, wave32, WMMA)
//   net = h0 @ Whh + bias + x @ Wih          (M=256, N=8192, K=4096, fp32)
//   thr = exact k-th largest |net| (k = 50%) via 2-pass radix select
//   gates -> c1, h1
// GEMM uses bf16x3 split on v_wmma_f32_16x16x32_bf16 (fp32-class accuracy,
// ~8/3x the K-rate of the native f32 WMMA; memory floor is 134MB/23.3TB/s).
// ---------------------------------------------------------------------------

typedef __attribute__((ext_vector_type(16))) __bf16 v16bf;
typedef __attribute__((ext_vector_type(8)))  __bf16 v8bf;
typedef __attribute__((ext_vector_type(4)))  __bf16 v4bf;
typedef __attribute__((ext_vector_type(8)))  float  v8f;

#define BATCH   256
#define KTOT    2048          // K per phase (x/Wih then h0/Whh)
#define N4H     8192
#define TILE_M  64
#define TILE_N  128
#define KC      32
#define LDA     40            // padded LDS row stride (bf16 elements)
#define NET_ELEMS (BATCH * N4H)             // 2097152
#define TOPK      (NET_ELEMS / 2)           // 1048576
#define OUT_HALF  (BATCH * 2048)            // 524288

// ---------------------------------------------------------------------------
// GEMM: net[256][8192] = x@Wih + h0@Whh + bias, bf16x3 WMMA
// grid = (64, 4), block = 256 (8 waves); wave computes 32x32 (2x2 WMMA tiles)
// ---------------------------------------------------------------------------
__global__ __launch_bounds__(256)
void sparse_lstm_gemm(const float* __restrict__ x,  const float* __restrict__ h0,
                      const float* __restrict__ Wih, const float* __restrict__ Whh,
                      const float* __restrict__ bias, float* __restrict__ net)
{
    __shared__ __bf16 sAhi[TILE_M * LDA];
    __shared__ __bf16 sAlo[TILE_M * LDA];
    __shared__ __bf16 sBhi[TILE_N * LDA];   // stored transposed: [n][k]
    __shared__ __bf16 sBlo[TILE_N * LDA];

    const int t     = threadIdx.x;
    const int lane  = t & 31;
    const int wave  = t >> 5;
    const int tileM = blockIdx.y * TILE_M;
    const int tileN = blockIdx.x * TILE_N;
    const int waveM = (wave & 1) * 32;
    const int waveN = (wave >> 1) * 32;

    v8f acc[2][2];
    #pragma unroll
    for (int i = 0; i < 2; ++i)
        #pragma unroll
        for (int j = 0; j < 2; ++j)
            #pragma unroll
            for (int e = 0; e < 8; ++e)
                acc[i][j][e] = 0.0f;

    const int rowL = lane & 15;
    const int k0a  = (lane < 16) ? 0 : 16;   // A frag: K runs {k0a..k0a+7, k0a+8..k0a+15}
    const int k0b  = (lane < 16) ? 0 : 16;   // B frag: K runs {k0b..k0b+7, k0b+8..k0b+15}

    for (int phase = 0; phase < 2; ++phase) {
        const float* __restrict__ Ag = phase ? h0  : x;
        const float* __restrict__ Wg = phase ? Whh : Wih;

        #pragma unroll 1
        for (int kb = 0; kb < KTOT; kb += KC) {
            // ---- stage A tile: 64 x 32 fp32 -> bf16 hi/lo in LDS ----
            #pragma unroll
            for (int g = 0; g < 2; ++g) {
                int idx = g * 256 + t;                 // 0..511
                int row = idx >> 3;                    // 0..63
                int k4  = (idx & 7) << 2;              // 0,4,..,28
                float4 a4 = *(const float4*)&Ag[(tileM + row) * 2048 + kb + k4];
                float av[4] = {a4.x, a4.y, a4.z, a4.w};
                v4bf hv, lv;
                #pragma unroll
                for (int c = 0; c < 4; ++c) {
                    __bf16 hb = (__bf16)av[c];
                    hv[c] = hb;
                    lv[c] = (__bf16)(av[c] - (float)hb);
                }
                *(v4bf*)&sAhi[row * LDA + k4] = hv;
                *(v4bf*)&sAlo[row * LDA + k4] = lv;
            }
            // ---- stage B tile: 32 x 128 fp32 -> transposed bf16 hi/lo ----
            #pragma unroll
            for (int g = 0; g < 4; ++g) {
                int idx = g * 256 + t;                 // 0..1023
                int n   = idx & 127;
                int k4  = (idx >> 7) << 2;             // 0,4,..,28
                v4bf hv, lv;
                #pragma unroll
                for (int c = 0; c < 4; ++c) {
                    float w = Wg[(kb + k4 + c) * N4H + tileN + n];  // coalesced over n
                    __bf16 hb = (__bf16)w;
                    hv[c] = hb;
                    lv[c] = (__bf16)(w - (float)hb);
                }
                *(v4bf*)&sBhi[n * LDA + k4] = hv;
                *(v4bf*)&sBlo[n * LDA + k4] = lv;
            }
            __syncthreads();

            // prefetch next K tile into cache while WMMAs run
            if (kb + KC < KTOT) {
                __builtin_prefetch(&Ag[(tileM + (t >> 3)) * 2048 + kb + KC], 0, 1);
                __builtin_prefetch(&Wg[(kb + KC + ((t >> 7) << 2)) * N4H + tileN + (t & 127)], 0, 1);
            }

            #pragma unroll
            for (int i = 0; i < 2; ++i) {
                int ar = (waveM + i * 16 + rowL) * LDA;
                v8bf a0h = *(const v8bf*)&sAhi[ar + k0a];
                v8bf a1h = *(const v8bf*)&sAhi[ar + k0a + 8];
                v8bf a0l = *(const v8bf*)&sAlo[ar + k0a];
                v8bf a1l = *(const v8bf*)&sAlo[ar + k0a + 8];
                v16bf ahi = __builtin_shufflevector(a0h, a1h, 0,1,2,3,4,5,6,7,8,9,10,11,12,13,14,15);
                v16bf alo = __builtin_shufflevector(a0l, a1l, 0,1,2,3,4,5,6,7,8,9,10,11,12,13,14,15);
                #pragma unroll
                for (int j = 0; j < 2; ++j) {
                    int br = (waveN + j * 16 + rowL) * LDA;
                    v8bf b0h = *(const v8bf*)&sBhi[br + k0b];
                    v8bf b1h = *(const v8bf*)&sBhi[br + k0b + 8];
                    v8bf b0l = *(const v8bf*)&sBlo[br + k0b];
                    v8bf b1l = *(const v8bf*)&sBlo[br + k0b + 8];
                    v16bf bhi = __builtin_shufflevector(b0h, b1h, 0,1,2,3,4,5,6,7,8,9,10,11,12,13,14,15);
                    v16bf blo = __builtin_shufflevector(b0l, b1l, 0,1,2,3,4,5,6,7,8,9,10,11,12,13,14,15);
                    // bf16x3: hi*hi + hi*lo + lo*hi, fp32 accumulate
                    acc[i][j] = __builtin_amdgcn_wmma_f32_16x16x32_bf16(
                        false, ahi, false, bhi, (short)0, acc[i][j], false, false);
                    acc[i][j] = __builtin_amdgcn_wmma_f32_16x16x32_bf16(
                        false, ahi, false, blo, (short)0, acc[i][j], false, false);
                    acc[i][j] = __builtin_amdgcn_wmma_f32_16x16x32_bf16(
                        false, alo, false, bhi, (short)0, acc[i][j], false, false);
                }
            }
            __syncthreads();
        }
    }

    // ---- epilogue: add bias, store net (C layout: VGPR r -> M=r / r+8) ----
    const int rOff = (lane < 16) ? 0 : 8;
    #pragma unroll
    for (int j = 0; j < 2; ++j) {
        int gn = tileN + waveN + j * 16 + rowL;
        float bv = bias[gn];
        #pragma unroll
        for (int i = 0; i < 2; ++i) {
            int gmBase = tileM + waveM + i * 16 + rOff;
            #pragma unroll
            for (int r = 0; r < 8; ++r)
                net[(gmBase + r) * N4H + gn] = acc[i][j][r] + bv;
        }
    }
}

// ---------------------------------------------------------------------------
// Radix-select support: histograms over abs() float bit patterns (monotonic)
// ---------------------------------------------------------------------------
__global__ void zero_hist_kernel(unsigned* __restrict__ p)
{
    p[blockIdx.x * 256 + threadIdx.x] = 0u;
}

__global__ void hist_hi_kernel(const float* __restrict__ net, unsigned* __restrict__ hist)
{
    int i = blockIdx.x * 256 + threadIdx.x;          // exactly NET_ELEMS threads
    unsigned bits = __float_as_uint(net[i]) & 0x7fffffffu;
    atomicAdd(&hist[bits >> 16], 1u);
}

__global__ void hist_lo_kernel(const float* __restrict__ net, const unsigned* __restrict__ sel,
                               unsigned* __restrict__ hist)
{
    int i = blockIdx.x * 256 + threadIdx.x;
    unsigned hb = sel[0];
    unsigned bits = __float_as_uint(net[i]) & 0x7fffffffu;
    if ((bits >> 16) == hb)
        atomicAdd(&hist[bits & 0xffffu], 1u);
}

// Find bin b (descending) with cum(>b) < K <= cum(>=b) over a 65536-bin hist.
// Launched with 256 threads; block-parallel partial sums, short serial tail.
__device__ __forceinline__ void radix_select_64k(const unsigned* __restrict__ hist,
                                                 unsigned K, unsigned* binOut, unsigned* remOut)
{
    __shared__ unsigned bsum[256];
    int t = threadIdx.x;
    unsigned s = 0;
    const uint4* hp = (const uint4*)(hist + t * 256);
    for (int i = 0; i < 64; ++i) {
        uint4 v = hp[i];
        s += v.x + v.y + v.z + v.w;
    }
    bsum[t] = s;
    __syncthreads();
    if (t == 0) {
        unsigned long long cum = 0;
        int blk = -1; unsigned rem = 0;
        for (int b = 255; b >= 0; --b) {
            unsigned c = bsum[b];
            if (cum + c >= (unsigned long long)K) { blk = b; rem = (unsigned)(K - cum); break; }
            cum += c;
        }
        if (blk < 0) { blk = 0; rem = 0; }
        unsigned long long c2 = 0;
        int bin = blk * 256;
        for (int i = 255; i >= 0; --i) {
            unsigned c = hist[blk * 256 + i];
            if (c2 + c >= (unsigned long long)rem) { bin = blk * 256 + i; rem = (unsigned)(rem - c2); break; }
            c2 += c;
        }
        *binOut = (unsigned)bin;
        *remOut = rem;
    }
}

__global__ void select_hi_kernel(const unsigned* __restrict__ hist, unsigned* __restrict__ sel)
{
    unsigned b = 0, r = 0;
    radix_select_64k(hist, (unsigned)TOPK, &b, &r);
    if (threadIdx.x == 0) { sel[0] = b; sel[1] = r; }
}

__global__ void select_lo_kernel(const unsigned* __restrict__ hist, unsigned* __restrict__ sel)
{
    unsigned K = sel[1];
    unsigned b = 0, r = 0;
    radix_select_64k(hist, K, &b, &r);
    if (threadIdx.x == 0) sel[2] = (sel[0] << 16) | b;   // exact k-th largest |net| bits
}

// ---------------------------------------------------------------------------
// Pointwise LSTM: mask (|v| > thr), gates, write h1 then c1
// ---------------------------------------------------------------------------
__global__ void lstm_pointwise_kernel(const float* __restrict__ net, const float* __restrict__ c0,
                                      const unsigned* __restrict__ sel, float* __restrict__ out)
{
    int i = blockIdx.x * 256 + threadIdx.x;          // 0..OUT_HALF-1
    float thr = __uint_as_float(sel[2]);
    int m = i >> 11;
    int col = i & 2047;
    const float* row = net + m * N4H;
    float f  = row[col];
    float ig = row[2048 + col];
    float o  = row[4096 + col];
    float g  = row[6144 + col];
    f  = (fabsf(f)  > thr) ? f  : 0.0f;
    ig = (fabsf(ig) > thr) ? ig : 0.0f;
    o  = (fabsf(o)  > thr) ? o  : 0.0f;
    g  = (fabsf(g)  > thr) ? g  : 0.0f;
    float sf = 1.0f / (1.0f + expf(-f));
    float si = 1.0f / (1.0f + expf(-ig));
    float so = 1.0f / (1.0f + expf(-o));
    float c1 = sf * c0[i] + si * tanhf(g);
    float h1 = so * tanhf(c1);
    out[i]            = h1;
    out[OUT_HALF + i] = c1;
}

// ---------------------------------------------------------------------------
extern "C" void kernel_launch(void* const* d_in, const int* in_sizes, int n_in,
                              void* d_out, int out_size, void* d_ws, size_t ws_size,
                              hipStream_t stream)
{
    (void)in_sizes; (void)n_in; (void)out_size; (void)ws_size;
    const float* x    = (const float*)d_in[0];
    const float* h0   = (const float*)d_in[1];
    const float* c0   = (const float*)d_in[2];
    const float* Wih  = (const float*)d_in[3];
    const float* Whh  = (const float*)d_in[4];
    const float* bias = (const float*)d_in[5];
    float* out = (float*)d_out;

    char* ws = (char*)d_ws;
    float*    net   = (float*)ws;                                  // 8 MB
    unsigned* hist1 = (unsigned*)(ws + (size_t)NET_ELEMS * 4);     // 256 KB
    unsigned* hist2 = hist1 + 65536;                               // 256 KB
    unsigned* sel   = hist2 + 65536;                               // few words

    zero_hist_kernel<<<512, 256, 0, stream>>>(hist1);              // zeros both hists (131072 u32)
    sparse_lstm_gemm<<<dim3(N4H / TILE_N, BATCH / TILE_M), 256, 0, stream>>>(
        x, h0, Wih, Whh, bias, net);
    hist_hi_kernel<<<NET_ELEMS / 256, 256, 0, stream>>>(net, hist1);
    select_hi_kernel<<<1, 256, 0, stream>>>(hist1, sel);
    hist_lo_kernel<<<NET_ELEMS / 256, 256, 0, stream>>>(net, sel, hist2);
    select_lo_kernel<<<1, 256, 0, stream>>>(hist2, sel);
    lstm_pointwise_kernel<<<OUT_HALF / 256, 256, 0, stream>>>(net, c0, sel, out);
}
